// EnhancedBeatFC_5111011082871
// MI455X (gfx1250) — matrix operand order repaired and checked by reading, not compiled, not run on hardware
//
#include <hip/hip_runtime.h>
#include <hip/hip_bf16.h>

#define B_   8
#define T_   4096
#define NMEL 26
#define WF   32
#define CCH  64
#define SEM  16
#define H1N  256
#define H2N  128
#define SHW  8
#define SHN  128
#define KLONG (WF * CCH)   // 2048
#define KSHORT (SHW * CCH) // 512
#define MT   32            // tokens per workgroup
#define KQ   512           // K quarter staged in LDS

typedef __attribute__((ext_vector_type(16))) __bf16 v16bf;
typedef __attribute__((ext_vector_type(8)))  float  v8f;

__device__ __forceinline__ unsigned short f2bf(float f) {
  union { float f; unsigned u; } v; v.f = f;
  unsigned r = v.u + 0x7FFFu + ((v.u >> 16) & 1u);
  return (unsigned short)(r >> 16);
}
__device__ __forceinline__ float bf2f(unsigned short h) {
  union { unsigned u; float f; } v; v.u = ((unsigned)h) << 16;
  return v.f;
}
// scale a packed bf16 pair by two fp32 factors, repack
__device__ __forceinline__ unsigned pack2(unsigned xpair, float s0, float s1) {
  float lo = bf2f((unsigned short)(xpair & 0xFFFFu)) * s0;
  float hi = bf2f((unsigned short)(xpair >> 16)) * s1;
  return (unsigned)f2bf(lo) | ((unsigned)f2bf(hi) << 16);
}

// ---------------- Stage 1: causal conv (26 -> 64) + ReLU, output bf16 ----------------
__global__ __launch_bounds__(256) void conv_kernel(
    const float* __restrict__ x, const float* __restrict__ cw,
    const float* __restrict__ cb, unsigned short* __restrict__ xc) {
  size_t idx = (size_t)blockIdx.x * 256 + threadIdx.x;   // B*T*64
  if (idx >= (size_t)B_ * T_ * CCH) return;
  int oc = (int)(idx & (CCH - 1));
  size_t bt = idx >> 6;
  int t = (int)(bt & (T_ - 1));
  int b = (int)(bt >> 12);
  float acc = cb[oc];
  #pragma unroll
  for (int k = 0; k < 5; ++k) {
    int st = t + k - 4;
    if (st < 0) continue;
    const float* xr = x + ((size_t)b * T_ + st) * NMEL;
    const float* wr = cw + (size_t)k * NMEL * CCH + oc;
    #pragma unroll
    for (int m = 0; m < NMEL; ++m) acc += xr[m] * wr[(size_t)m * CCH];
  }
  acc = acc > 0.f ? acc : 0.f;
  xc[idx] = f2bf(acc);
}

// ---------------- Stage 2: SE attention per token ----------------
__global__ __launch_bounds__(256) void se_kernel(
    const unsigned short* __restrict__ xc,
    const float* __restrict__ sw1, const float* __restrict__ sb1,
    const float* __restrict__ sw2, const float* __restrict__ sb2,
    float* __restrict__ a_out) {
  __shared__ float s[4][CCH];
  __shared__ float mid[4][SEM];
  int tid = threadIdx.x;
  int tr = tid >> 6;           // 0..3 tokens per block
  int c  = tid & 63;
  size_t gt = (size_t)blockIdx.x * 4 + tr;
  int b = (int)(gt >> 12);
  int t = (int)(gt & (T_ - 1));
  float sum = 0.f;
  #pragma unroll 4
  for (int f = 0; f < WF; ++f) {
    int st = t - (WF - 1) + f;
    if (st >= 0) sum += bf2f(xc[((size_t)b * T_ + st) * CCH + c]);
  }
  s[tr][c] = sum * (1.0f / 32.0f);
  __syncthreads();
  if (tid < 64) {
    int r = tid >> 4, j = tid & 15;
    float m = sb1[j];
    #pragma unroll 8
    for (int cc = 0; cc < CCH; ++cc) m += s[r][cc] * sw1[cc * SEM + j];
    mid[r][j] = m > 0.f ? m : 0.f;
  }
  __syncthreads();
  float att = sb2[c];
  #pragma unroll
  for (int j = 0; j < SEM; ++j) att += mid[tr][j] * sw2[j * CCH + c];
  att = 1.0f / (1.0f + __expf(-att));
  a_out[((size_t)b * T_ + t) * CCH + c] = att;
}

// ---------------- Stage 3: weight transpose + fp32->bf16 ----------------
__global__ __launch_bounds__(256) void prep_kernel(
    const float* __restrict__ w1, const float* __restrict__ w2,
    const float* __restrict__ sw,
    unsigned short* __restrict__ w1T, unsigned short* __restrict__ w2T,
    unsigned short* __restrict__ swT) {
  int idx = blockIdx.x * 256 + threadIdx.x;
  if (idx < KLONG * H1N) {
    int k = idx / H1N, n = idx % H1N;
    w1T[(size_t)n * KLONG + k] = f2bf(w1[idx]);
  }
  if (idx < H1N * H2N) {
    int k = idx / H2N, n = idx % H2N;
    w2T[(size_t)n * H1N + k] = f2bf(w2[idx]);
  }
  if (idx < KSHORT * SHN) {
    int k = idx / SHN, n = idx % SHN;
    swT[(size_t)n * KSHORT + k] = f2bf(sw[idx]);
  }
}

// Fragment pack: two A tiles (token rows 0-15 / 16-31) + one shared B tile.
// A frag (16x32, ISA 7.12.2): hk=(lane>=16)?8:0; V0..3 = K pairs hk+2v; V4..7 = 16+hk+2v.
// B frag (32x16): same K layout as A: V0..3 = K pairs hk+2v; V4..7 = 16+hk+2v.
struct FragAB {
  union { v16bf v; unsigned u[8]; } a0, a1, b;
};

__device__ __forceinline__ void load_ab(FragAB& f,
                                        const unsigned* __restrict__ arow0,
                                        const unsigned* __restrict__ arow1,
                                        const unsigned* __restrict__ brow,
                                        int baseDw, int hk2, int kb2) {
  #pragma unroll
  for (int v = 0; v < 4; ++v) {
    f.a0.u[v]     = arow0[baseDw + hk2 + v];
    f.a0.u[v + 4] = arow0[baseDw + 8 + hk2 + v];
    f.a1.u[v]     = arow1[baseDw + hk2 + v];
    f.a1.u[v + 4] = arow1[baseDw + 8 + hk2 + v];
  }
  #pragma unroll
  for (int v = 0; v < 8; ++v) f.b.u[v] = brow[baseDw + hk2 + (v & 3) + ((v >> 2) << 3)];
}
__device__ __forceinline__ void mma2(const FragAB& f, v8f& accA, v8f& accB) {
  accA = __builtin_amdgcn_wmma_f32_16x16x32_bf16(false, f.a0.v, false, f.b.v,
                                                 (short)0, accA, false, false);
  accB = __builtin_amdgcn_wmma_f32_16x16x32_bf16(false, f.a1.v, false, f.b.v,
                                                 (short)0, accB, false, false);
}

// Double-buffered K loop: next fragments in flight while current WMMAs issue.
__device__ __forceinline__ void gemm_loop(const unsigned* __restrict__ arow0,
                                          const unsigned* __restrict__ arow1,
                                          const unsigned* __restrict__ brow,
                                          int nkb, int lane, v8f& accA, v8f& accB) {
  int hk2 = (lane >> 4) * 4;
  int kb2 = (lane >> 4) * 8;
  FragAB f0, f1;
  load_ab(f0, arow0, arow1, brow, 0, hk2, kb2);
  for (int kb = 0; kb < nkb; kb += 2) {   // nkb is even in all uses
    load_ab(f1, arow0, arow1, brow, (kb + 1) * 16, hk2, kb2);
    mma2(f0, accA, accB);
    if (kb + 2 < nkb) load_ab(f0, arow0, arow1, brow, (kb + 2) * 16, hk2, kb2);
    mma2(f1, accA, accB);
  }
}

// ---------------- Stage 4: fused implicit-GEMM pipeline ----------------
// Each workgroup: 32 tokens, 512 threads = 16 waves. LDS = 32KB A-tile + 16KB H buffer.
__global__ __launch_bounds__(512) void fused_kernel(
    const unsigned short* __restrict__ xc,   // [B][T][64] bf16
    const float* __restrict__ a,             // [B][T][64] f32
    const unsigned short* __restrict__ w1T,  // [256][2048] bf16
    const float* __restrict__ b1,
    const unsigned short* __restrict__ w2T,  // [128][256] bf16
    const float* __restrict__ b2,
    const unsigned short* __restrict__ swT,  // [128][512] bf16
    const float* __restrict__ sb,
    const float* __restrict__ wo,            // [256][2] f32
    const float* __restrict__ bo,
    float* __restrict__ out) {               // [B][T][2] f32
  __shared__ unsigned short At[MT][KQ];      // 32KB: quarter-K scaled window tile
  __shared__ unsigned short Hbuf[MT][H1N];   // 16KB: H1, then reused for concat H2

  int wg  = blockIdx.x;                 // B*T/32 workgroups
  int b   = wg >> 7;                    // T/32 = 128 WGs per batch
  int t0  = (wg & 127) << 5;
  int tid = threadIdx.x;
  int lane = tid & 31;
  int wv   = tid >> 5;                  // 0..15
  int r16  = lane & 15;

  v8f accA = {}, accB = {};
  int n0 = wv * 16;                     // this wave's N tile of w1 (256 cols)

  #pragma unroll 1
  for (int q = 0; q < 4; ++q) {
    if (q) __syncthreads();             // readers of previous quarter done
    // ---- cooperative fill: 32 rows x 512 K; 512 threads, 32 K each (vectorized) ----
    {
      int row   = tid >> 4;             // 0..31
      int sub   = tid & 15;
      int token = t0 + row;
      int kLoc  = sub * 32;             // 32 contiguous K = half a frame's channels
      int kGlob = q * KQ + kLoc;
      int f     = kGlob >> 6;
      int c0    = kGlob & 63;
      int srcT  = token - (WF - 1) + f;
      uint4* dst = (uint4*)&At[row][kLoc];
      if (srcT >= 0) {
        const uint4*  xr = (const uint4*)(xc + ((size_t)b * T_ + srcT) * CCH + c0);
        const float4* ar = (const float4*)(a + ((size_t)b * T_ + token) * CCH + c0);
        #pragma unroll
        for (int i = 0; i < 4; ++i) {
          uint4 xv = xr[i];
          float4 s0 = ar[2 * i], s1 = ar[2 * i + 1];
          uint4 o;
          o.x = pack2(xv.x, s0.x, s0.y);
          o.y = pack2(xv.y, s0.z, s0.w);
          o.z = pack2(xv.z, s1.x, s1.y);
          o.w = pack2(xv.w, s1.z, s1.w);
          dst[i] = o;
        }
      } else {
        uint4 z = make_uint4(0, 0, 0, 0);
        #pragma unroll
        for (int i = 0; i < 4; ++i) dst[i] = z;
      }
    }
    __syncthreads();
    // ---- 16 WMMA K-steps over this quarter (2 wmma each, shared B frag) ----
    const unsigned* arow0 = (const unsigned*)&At[r16][0];
    const unsigned* arow1 = (const unsigned*)&At[16 + r16][0];
    const unsigned* brow =
        (const unsigned*)(w1T + (size_t)(n0 + r16) * KLONG) + q * (KQ / 2);
    gemm_loop(arow0, arow1, brow, 16, lane, accA, accB);
  }

  // ---- bias + relu -> H1 bf16 (D layout: M = r + 8*(lane>=16), N = lane&15) ----
  {
    int mh = (lane >> 4) * 8;
    int n  = n0 + r16;
    float bias = b1[n];
    #pragma unroll
    for (int r = 0; r < 8; ++r) {
      float vA = accA[r] + bias;
      float vB = accB[r] + bias;
      Hbuf[mh + r][n]      = f2bf(vA > 0.f ? vA : 0.f);
      Hbuf[16 + mh + r][n] = f2bf(vB > 0.f ? vB : 0.f);
    }
  }
  __syncthreads();
  // At still holds quarter 3 = global K 1536..2047 = exactly the short window.

  v8f acc2A = {}, acc2B = {};
  int n2 = (wv & 7) * 16;
  if (wv < 8) {
    // h_short = relu(shortWindow(512) @ sw) : waves 0..7, N tiles of 128
    const unsigned* arow0 = (const unsigned*)&At[r16][0];
    const unsigned* arow1 = (const unsigned*)&At[16 + r16][0];
    const unsigned* brow  = (const unsigned*)(swT + (size_t)(n2 + r16) * KSHORT);
    gemm_loop(arow0, arow1, brow, 16, lane, acc2A, acc2B);
  } else {
    // h2 = relu(h1(256) @ w2) : waves 8..15, N tiles of 128
    const unsigned* arow0 = (const unsigned*)&Hbuf[r16][0];
    const unsigned* arow1 = (const unsigned*)&Hbuf[16 + r16][0];
    const unsigned* brow  = (const unsigned*)(w2T + (size_t)(n2 + r16) * H1N);
    gemm_loop(arow0, arow1, brow, 8, lane, acc2A, acc2B);
  }
  __syncthreads();   // all reads of H1 (and At) complete before Hbuf is reused

  // store into concat layout: h2 at cols [0,128), h_short at cols [128,256)
  {
    int mh = (lane >> 4) * 8;
    int n  = n2 + r16;
    float bias = (wv < 8) ? sb[n] : b2[n];
    int   col  = (wv < 8) ? (128 + n) : n;
    #pragma unroll
    for (int r = 0; r < 8; ++r) {
      float vA = acc2A[r] + bias;
      float vB = acc2B[r] + bias;
      Hbuf[mh + r][col]      = f2bf(vA > 0.f ? vA : 0.f);
      Hbuf[16 + mh + r][col] = f2bf(vB > 0.f ? vB : 0.f);
    }
  }
  __syncthreads();

  // ---- output head: 32 tokens x 2, sigmoid ----
  if (tid < 64) {
    int m = tid >> 1, o = tid & 1;
    float s = bo[o];
    #pragma unroll 8
    for (int k = 0; k < 256; ++k) s += bf2f(Hbuf[m][k]) * wo[k * 2 + o];
    out[((size_t)b * T_ + t0 + m) * 2 + o] = 1.0f / (1.0f + __expf(-s));
  }
}

extern "C" void kernel_launch(void* const* d_in, const int* in_sizes, int n_in,
                              void* d_out, int out_size, void* d_ws, size_t ws_size,
                              hipStream_t stream) {
  (void)in_sizes; (void)n_in; (void)out_size; (void)ws_size;
  const float* x      = (const float*)d_in[0];
  const float* conv_w = (const float*)d_in[1];
  const float* conv_b = (const float*)d_in[2];
  const float* se_w1  = (const float*)d_in[3];
  const float* se_b1  = (const float*)d_in[4];
  const float* se_w2  = (const float*)d_in[5];
  const float* se_b2  = (const float*)d_in[6];
  const float* w1     = (const float*)d_in[7];
  const float* b1     = (const float*)d_in[8];
  const float* w2     = (const float*)d_in[9];
  const float* b2     = (const float*)d_in[10];
  const float* sw     = (const float*)d_in[11];
  const float* sb     = (const float*)d_in[12];
  const float* wo     = (const float*)d_in[13];
  const float* bo     = (const float*)d_in[14];
  float* out = (float*)d_out;

  // workspace layout
  char* ws = (char*)d_ws;
  unsigned short* xc  = (unsigned short*)(ws);                    // 4 MB
  float*          att = (float*)(ws + 4u * 1024 * 1024);          // 8 MB
  unsigned short* w1T = (unsigned short*)(ws + 12u * 1024 * 1024);// 1 MB
  unsigned short* w2T = (unsigned short*)(ws + 13u * 1024 * 1024);// 64 KB
  unsigned short* swT = (unsigned short*)(ws + 13u * 1024 * 1024 + 256u * 1024); // 128 KB

  conv_kernel<<<(B_ * T_ * CCH) / 256, 256, 0, stream>>>(x, conv_w, conv_b, xc);
  prep_kernel<<<(KLONG * H1N) / 256, 256, 0, stream>>>(w1, w2, sw, w1T, w2T, swT);
  se_kernel<<<(B_ * T_) / 4, 256, 0, stream>>>(xc, se_w1, se_b1, se_w2, se_b2, att);
  fused_kernel<<<(B_ * T_) / 32, 512, 0, stream>>>(
      xc, att, w1T, b1, w2T, b2, swT, sb, wo, bo, out);
}
